// HamiltonianNano_v8_90649579749476
// MI455X (gfx1250) — matrix operand order repaired and checked by reading, not compiled, not run on hardware
//
#include <hip/hip_runtime.h>
#include <hip/hip_bf16.h>

// ---------------------------------------------------------------------------
// HamiltonianNano forward for MI455X (gfx1250, wave32, WMMA bf16).
// All GEMMs + attention use V_WMMA_F32_16X16X32_BF16 (fp32 accumulate).
// GEMM stages weight tiles through LDS (shared by all 8 waves of a block);
// flash attention uses a far/near split for the gravity-distance bias and
// batches all fragment loads ahead of the WMMA group to hide load latency.
// ---------------------------------------------------------------------------

typedef __bf16 bf16;
typedef bf16  bf16x16 __attribute__((ext_vector_type(16)));
typedef float v8f     __attribute__((ext_vector_type(8)));

union Frag {            // 16 bf16 = 32 bytes = two 16B chunks
    bf16x16 v;
    uint4   q[2];
};

__device__ __forceinline__ v8f wmma_bf16(bf16x16 a, bf16x16 b, v8f c) {
    // D(16x16 f32) = A(16x32 bf16) * B(32x16 bf16) + C
    return __builtin_amdgcn_wmma_f32_16x16x32_bf16(
        /*neg_a=*/false, a, /*neg_b=*/false, b,
        /*c_mod=*/(short)0, c, /*reuse_a=*/false, /*reuse_b=*/false);
}

__device__ __forceinline__ bf16 f2bf(float f) {
    unsigned u = __float_as_uint(f);
    unsigned r = (u + 0x7FFFu + ((u >> 16) & 1u)) >> 16;   // RNE
    unsigned short s = (unsigned short)r;
    return __builtin_bit_cast(bf16, s);
}
__device__ __forceinline__ float bf2f(bf16 b) {
    unsigned short s = __builtin_bit_cast(unsigned short, b);
    return __uint_as_float(((unsigned)s) << 16);
}
__device__ __forceinline__ float gelu_f(float v) {
    return 0.5f * v * (1.0f + erff(v * 0.70710678118654752f));
}

// Model constants
#define CB 2
#define CT 2048
#define CC 768
#define CH 12
#define CD 64
#define CV 50257
#define CL 6
#define CK 4

// ---------------------------------------------------------------------------
// fp32 -> bf16 conversion (weights, once per launch)
// ---------------------------------------------------------------------------
__global__ void cvt_bf16_k(const float* __restrict__ in, bf16* __restrict__ out, size_t n) {
    size_t i = (size_t)blockIdx.x * 256 + threadIdx.x;
    if (i < n) out[i] = f2bf(in[i]);
}

// ---------------------------------------------------------------------------
// Embedding gather + EMA momentum scan: x = q_emb[idx] + scan(beta, p-diff)
// ---------------------------------------------------------------------------
__global__ void embed_ema_k(const int* __restrict__ idx, const float* __restrict__ qe,
                            const float* __restrict__ pe, float* __restrict__ x) {
    int i = blockIdx.x * 256 + threadIdx.x;
    if (i >= CB * CC) return;
    int c = i % CC, b = i / CC;
    float m = 0.f, pprev = 0.f;
    for (int t = 0; t < CT; ++t) {
        int id = idx[b * CT + t];
        id = id < 0 ? 0 : (id > CV - 1 ? CV - 1 : id);
        float p = pe[(size_t)id * CC + c];
        float vel = (t == 0) ? 0.f : (p - pprev);
        m = 0.95f * m + 0.05f * vel;
        x[((size_t)b * CT + t) * CC + c] = qe[(size_t)id * CC + c] + m;
        pprev = p;
    }
}

// ---------------------------------------------------------------------------
// Depthwise causal conv (K=4) + SiLU on raw x -> reflex
// ---------------------------------------------------------------------------
__global__ void conv_silu_k(const float* __restrict__ x, const float* __restrict__ cw,
                            const float* __restrict__ cb, float* __restrict__ out) {
    size_t i = (size_t)blockIdx.x * 256 + threadIdx.x;
    if (i >= (size_t)CB * CT * CC) return;
    int c = (int)(i % CC);
    size_t bt = i / CC;
    int t = (int)(bt % CT);
    int b = (int)(bt / CT);
    float acc = cb[c];
#pragma unroll
    for (int k = 0; k < CK; ++k) {
        int tt = t - (CK - 1) + k;
        if (tt >= 0) acc += cw[c * CK + k] * x[((size_t)b * CT + tt) * CC + c];
    }
    out[i] = acc / (1.f + __expf(-acc));   // silu
}

// ---------------------------------------------------------------------------
// LayerNorm over C=768; one block (256 thr) per row; bf16 out (+opt fp32 out)
// ---------------------------------------------------------------------------
template <bool WF32>
__global__ void __launch_bounds__(256)
layernorm_k(const float* __restrict__ x, const float* __restrict__ w,
            const float* __restrict__ bias, bf16* __restrict__ ob, float* __restrict__ of) {
    int row = blockIdx.x;
    const float* xr = x + (size_t)row * CC;
    float vals[3], s = 0.f, ss = 0.f;
#pragma unroll
    for (int i = 0; i < 3; ++i) {
        float v = xr[threadIdx.x + 256 * i];
        vals[i] = v; s += v; ss += v * v;
    }
    for (int m = 1; m < 32; m <<= 1) { s += __shfl_xor(s, m); ss += __shfl_xor(ss, m); }
    __shared__ float sh[16];
    int wave = threadIdx.x >> 5, lane = threadIdx.x & 31;
    if (lane == 0) { sh[wave] = s; sh[8 + wave] = ss; }
    __syncthreads();
    if (threadIdx.x == 0) {
        float ts = 0.f, tss = 0.f;
        for (int i = 0; i < 8; ++i) { ts += sh[i]; tss += sh[8 + i]; }
        float mean = ts / (float)CC;
        float var = tss / (float)CC - mean * mean;
        sh[0] = mean; sh[1] = rsqrtf(var + 1e-5f);
    }
    __syncthreads();
    float mean = sh[0], rinv = sh[1];
#pragma unroll
    for (int i = 0; i < 3; ++i) {
        int c = threadIdx.x + 256 * i;
        float v = (vals[i] - mean) * rinv * w[c] + bias[c];
        ob[(size_t)row * CC + c] = f2bf(v);
        if (WF32) of[(size_t)row * CC + c] = v;
    }
}

// ---------------------------------------------------------------------------
// WMMA GEMM with LDS-staged weights: C[m,n] = act( sum_k A[m,k]*B[n,k] )
// A bf16 [M,K] row-major, B bf16 [N,K] row-major (computes A @ B^T).
// Block: 256 thr = 8 waves, block tile 256x64. The 64x64 B tile per k-step is
// staged once in LDS and shared by all waves. Each wave computes 32x64 via
// 2 A-fragments x 4 B-fragments -> 8 WMMA per 32-k step.
// ACT: 0=none, 1=exact gelu. OUT_BF16 selects destination type.
// ---------------------------------------------------------------------------
template <int ACT, bool OUT_BF16>
__global__ void __launch_bounds__(256)
gemm_bf16_k(const bf16* __restrict__ A, const bf16* __restrict__ Bw,
            float* __restrict__ Cf, bf16* __restrict__ Cb, int M, int N, int Kd) {
    __shared__ bf16 Bs[64][72];            // 64 n-rows x 64 k (+8 pad), 9.2 KB
    int wave = threadIdx.x >> 5, lane = threadIdx.x & 31;
    int hi = lane >> 4, ln16 = lane & 15;
    int m0 = (blockIdx.x * 8 + wave) * 32;
    int n0 = blockIdx.y * 64;

    v8f acc[2][4];
#pragma unroll
    for (int mi = 0; mi < 2; ++mi)
#pragma unroll
        for (int j = 0; j < 4; ++j)
#pragma unroll
            for (int r = 0; r < 8; ++r) acc[mi][j][r] = 0.f;

    int ra0 = m0 + ln16;      if (ra0 > M - 1) ra0 = M - 1;
    int ra1 = m0 + 16 + ln16; if (ra1 > M - 1) ra1 = M - 1;
    const bf16* arow0 = A + (size_t)ra0 * Kd;
    const bf16* arow1 = A + (size_t)ra1 * Kd;

    for (int k0 = 0; k0 < Kd; k0 += 64) {
        __syncthreads();                   // protect Bs readers of prev step
        // ---- cooperative stage of B tile: 64 rows x 64 k ----
#pragma unroll
        for (int it = 0; it < 2; ++it) {
            int li = threadIdx.x + it * 256;     // 0..511, 8 bf16 each
            int rowN = li >> 3;
            int colK = (li & 7) * 8;
            int n = n0 + rowN; if (n > N - 1) n = N - 1;
            *(uint4*)&Bs[rowN][colK] = *(const uint4*)(Bw + (size_t)n * Kd + k0 + colK);
        }
        __syncthreads();
        __builtin_prefetch(arow0 + k0 + 64, 0, 3);   // global_prefetch_b8
        __builtin_prefetch(arow1 + k0 + 64, 0, 3);
#pragma unroll
        for (int ks = 0; ks < 64; ks += 32) {
            // A fragments (16x32): lo-lanes K {0..7,16..23}, hi-lanes {8..15,24..31}
            Frag a0, a1;
            a0.q[0] = *(const uint4*)(arow0 + k0 + ks + hi * 8);
            a0.q[1] = *(const uint4*)(arow0 + k0 + ks + 16 + hi * 8);
            a1.q[0] = *(const uint4*)(arow1 + k0 + ks + hi * 8);
            a1.q[1] = *(const uint4*)(arow1 + k0 + ks + 16 + hi * 8);
            // B fragments (32x16): lane col = lane&15; lo K {0..7,16..23}, hi K {8..15,24..31}
            Frag bf_[4];
#pragma unroll
            for (int j = 0; j < 4; ++j) {
                const bf16* bp = &Bs[j * 16 + ln16][ks + hi * 8];
                bf_[j].q[0] = *(const uint4*)bp;
                bf_[j].q[1] = *(const uint4*)(bp + 16);
            }
#pragma unroll
            for (int j = 0; j < 4; ++j) {
                acc[0][j] = wmma_bf16(a0.v, bf_[j].v, acc[0][j]);
                acc[1][j] = wmma_bf16(a1.v, bf_[j].v, acc[1][j]);
            }
        }
    }
    // C layout: vgpr r, lane l -> row +r+(l>=16)*8, col j*16+(l&15)
#pragma unroll
    for (int mi = 0; mi < 2; ++mi)
#pragma unroll
        for (int j = 0; j < 4; ++j) {
            int col = n0 + j * 16 + ln16;
            if (col >= N) continue;
#pragma unroll
            for (int r = 0; r < 8; ++r) {
                int rowm = m0 + mi * 16 + r + hi * 8;
                if (rowm >= M) continue;
                float v = acc[mi][j][r];
                if (ACT == 1) v = gelu_f(v);
                if (OUT_BF16) Cb[(size_t)rowm * N + col] = f2bf(v);
                else          Cf[(size_t)rowm * N + col] = v;
            }
        }
}

// ---------------------------------------------------------------------------
// RoPE + split qkv[B,T,3C] -> Q,K bf16 [B,H,T,D] (rotated), V^T bf16 [B,H,D,T]
// ---------------------------------------------------------------------------
__global__ void rope_split_k(const float* __restrict__ qkv, bf16* __restrict__ qb,
                             bf16* __restrict__ kb, bf16* __restrict__ vt) {
    size_t i = (size_t)blockIdx.x * 256 + threadIdx.x;     // B*T*H*D
    if (i >= (size_t)CB * CT * CH * CD) return;
    int d = (int)(i & (CD - 1));
    size_t r = i >> 6;
    int h = (int)(r % CH);
    size_t bt = r / CH;
    int t = (int)(bt % CT);
    int b = (int)(bt / CT);
    size_t base = ((size_t)b * CT + t) * (3 * CC) + h * CD;
    float qv = qkv[base + d];
    float kv = qkv[base + CC + d];
    float vv = qkv[base + 2 * CC + d];
    int j = d & 31;
    float invf = __powf(10000.f, -(float)(2 * j) / (float)CD);
    float ang = (float)t * invf;
    float cs = __cosf(ang), sn = __sinf(ang);
    float q2, k2;
    if (d < 32) { q2 = qv * cs - qkv[base + d + 32] * sn;
                  k2 = kv * cs - qkv[base + CC + d + 32] * sn; }
    else        { q2 = qv * cs + qkv[base + d - 32] * sn;
                  k2 = kv * cs + qkv[base + CC + d - 32] * sn; }
    size_t bh = (size_t)b * CH + h;
    qb[(bh * CT + t) * CD + d] = f2bf(q2);
    kb[(bh * CT + t) * CD + d] = f2bf(k2);
    vt[(bh * CD + d) * CT + t] = f2bf(vv);
}

// ---------------------------------------------------------------------------
// Gravity head: G[b,h,t] = softplus( ln1(x)[b,t,:] . gw[h,:] + gb[h] )
// ---------------------------------------------------------------------------
__global__ void gravity_k(const bf16* __restrict__ xb, const float* __restrict__ gw,
                          const float* __restrict__ gb, float* __restrict__ G) {
    int i = blockIdx.x * 256 + threadIdx.x;    // B*T*H
    if (i >= CB * CT * CH) return;
    int h = i % CH;
    int bt = i / CH;
    int t = bt % CT, b = bt / CT;
    const bf16* xr = xb + (size_t)bt * CC;
    const float* wr = gw + h * CC;
    float acc = gb[h];
    for (int c = 0; c < CC; ++c) acc += bf2f(xr[c]) * wr[c];
    float sp = acc > 20.f ? acc : log1pf(__expf(acc));
    G[((size_t)b * CH + h) * CT + t] = sp;
}

// ---------------------------------------------------------------------------
// Flash attention with WMMA. One wave per 16-row Q tile; block = 4 waves
// covering 64 consecutive q rows of one (b,h); uniform k-loop (barriers are
// in uniform control flow, causality via masking). Far-field fast path:
// for k32 <= q0-95 every distance saturates at DIST_CAP=64, so the bias is a
// per-row constant and no causal mask / int converts are needed.
// All K / V fragments of a step are loaded before the WMMA group so the
// loads clause together and the waits are partial (latency overlap).
// ---------------------------------------------------------------------------
__global__ void __launch_bounds__(128)
flash_attn_k(const bf16* __restrict__ Q, const bf16* __restrict__ Km,
             const bf16* __restrict__ Vt, const float* __restrict__ G,
             bf16* __restrict__ O, float scale) {
    __shared__ bf16 pbuf[4][16 * 32];
    int wave = threadIdx.x >> 5, lane = threadIdx.x & 31;
    int hi = lane >> 4, ln16 = lane & 15;
    int blk = blockIdx.x;                 // B*H*(T/64)
    int qb64 = blk % (CT / 64);
    int bh = blk / (CT / 64);
    int b = bh / CH, h = bh % CH;
    int q0 = qb64 * 64 + wave * 16;

    const bf16* Qp = Q + ((size_t)bh * CT + q0) * CD;
    const bf16* Kp = Km + (size_t)bh * CT * CD;
    const bf16* Vp = Vt + (size_t)bh * CD * CT;
    bf16* pw = pbuf[wave];

    // Q fragments for d 0..31 and 32..63 (loaded once)
    Frag aq[2];
    const bf16* qrow = Qp + ln16 * CD;
    aq[0].q[0] = *(const uint4*)(qrow + hi * 8);
    aq[0].q[1] = *(const uint4*)(qrow + 16 + hi * 8);
    aq[1].q[0] = *(const uint4*)(qrow + 32 + hi * 8);
    aq[1].q[1] = *(const uint4*)(qrow + 48 + hi * 8);

    float gq01[8], gfar[8];
#pragma unroll
    for (int r = 0; r < 8; ++r) {
        float g = G[(size_t)bh * CT + q0 + r + hi * 8];
        gq01[r] = g * 0.1f;            // bias slope
        gfar[r] = g * 6.4f;            // saturated bias: g * 64 * 0.1
    }

    v8f o[4];
    float rmax[8], rsum[8];
#pragma unroll
    for (int j = 0; j < 4; ++j)
#pragma unroll
        for (int r = 0; r < 8; ++r) o[j][r] = 0.f;
#pragma unroll
    for (int r = 0; r < 8; ++r) { rmax[r] = -3e38f; rsum[r] = 0.f; }

    int kend = qb64 * 64 + 64;            // uniform across block
    for (int k32 = 0; k32 < kend; k32 += 32) {
        // ---- scores for two 16-col subtiles; load all 4 K frags first ----
        const bf16* krA = Kp + (size_t)(k32 + ln16) * CD;
        const bf16* krB = Kp + (size_t)(k32 + 16 + ln16) * CD;
        Frag kf[4];                        // [subtile(A/B)][d-chunk]
#pragma unroll
        for (int c = 0; c < 2; ++c) {
            kf[c].q[0]     = *(const uint4*)(krA + c * 32 + hi * 8);
            kf[c].q[1]     = *(const uint4*)(krA + c * 32 + hi * 8 + 16);
            kf[2 + c].q[0] = *(const uint4*)(krB + c * 32 + hi * 8);
            kf[2 + c].q[1] = *(const uint4*)(krB + c * 32 + hi * 8 + 16);
        }
        v8f sA, sB;
#pragma unroll
        for (int r = 0; r < 8; ++r) { sA[r] = 0.f; sB[r] = 0.f; }
        sA = wmma_bf16(aq[0].v, kf[0].v, sA);
        sB = wmma_bf16(aq[0].v, kf[2].v, sB);
        sA = wmma_bf16(aq[1].v, kf[1].v, sA);
        sB = wmma_bf16(aq[1].v, kf[3].v, sB);

        if (k32 <= q0 - 95) {
            // far field: dist saturated at 64 for every element of both subtiles
#pragma unroll
            for (int r = 0; r < 8; ++r) {
                sA[r] = sA[r] * scale - gfar[r];
                sB[r] = sB[r] * scale - gfar[r];
            }
        } else {
            int kca = k32 + ln16, kcb = k32 + 16 + ln16;
#pragma unroll
            for (int r = 0; r < 8; ++r) {
                int qrw = q0 + r + hi * 8;
                float da = fminf(fabsf((float)(qrw - kca)), 64.f);
                float db = fminf(fabsf((float)(qrw - kcb)), 64.f);
                float sa = sA[r] * scale - gq01[r] * da;
                float sb = sB[r] * scale - gq01[r] * db;
                if (kca > qrw) sa = -3e38f;
                if (kcb > qrw) sb = -3e38f;
                sA[r] = sa; sB[r] = sb;
            }
        }
        // ---- online softmax update (rows live across 16-lane halves) ----
#pragma unroll
        for (int r = 0; r < 8; ++r) {
            float mt = fmaxf(sA[r], sB[r]);
            for (int m = 1; m <= 8; m <<= 1) mt = fmaxf(mt, __shfl_xor(mt, m));
            float mnew = fmaxf(rmax[r], mt);
            float corr = __expf(rmax[r] - mnew);
            float pa = __expf(sA[r] - mnew);
            float pb = __expf(sB[r] - mnew);
            float ps = pa + pb;
            for (int m = 1; m <= 8; m <<= 1) ps += __shfl_xor(ps, m);
            rsum[r] = rsum[r] * corr + ps;
            rmax[r] = mnew;
#pragma unroll
            for (int j = 0; j < 4; ++j) o[j][r] *= corr;
            int rl = r + hi * 8;
            pw[rl * 32 + ln16] = f2bf(pa);
            pw[rl * 32 + 16 + ln16] = f2bf(pb);
        }
        __syncthreads();
        // ---- O += P(16x32) @ V(32x64); load all 4 V frags first ----
        Frag ap;
        ap.q[0] = *(const uint4*)(pw + ln16 * 32 + hi * 8);
        ap.q[1] = *(const uint4*)(pw + ln16 * 32 + 16 + hi * 8);
        Frag bv[4];
#pragma unroll
        for (int j = 0; j < 4; ++j) {
            const bf16* vr = Vp + (size_t)(j * 16 + ln16) * CT + k32 + hi * 8;
            bv[j].q[0] = *(const uint4*)vr;
            bv[j].q[1] = *(const uint4*)(vr + 16);
        }
#pragma unroll
        for (int j = 0; j < 4; ++j) o[j] = wmma_bf16(ap.v, bv[j].v, o[j]);
        __syncthreads();
    }
    // ---- normalize + store as bf16 [B,T,C] ----
#pragma unroll
    for (int j = 0; j < 4; ++j)
#pragma unroll
        for (int r = 0; r < 8; ++r) {
            int t = q0 + r + hi * 8;
            float val = o[j][r] / rsum[r];
            O[((size_t)b * CT + t) * CC + h * CD + j * 16 + ln16] = f2bf(val);
        }
}

// ---------------------------------------------------------------------------
// Residual adds
// ---------------------------------------------------------------------------
__global__ void add3_k(float* __restrict__ x, const float* __restrict__ a,
                       const float* __restrict__ c, size_t n) {
    size_t i = (size_t)blockIdx.x * 256 + threadIdx.x;
    if (i < n) x[i] += a[i] + c[i];
}
__global__ void add1_k(float* __restrict__ x, const float* __restrict__ a, size_t n) {
    size_t i = (size_t)blockIdx.x * 256 + threadIdx.x;
    if (i < n) x[i] += a[i];
}

// ---------------------------------------------------------------------------
extern "C" void kernel_launch(void* const* d_in, const int* in_sizes, int n_in,
                              void* d_out, int out_size, void* d_ws, size_t ws_size,
                              hipStream_t stream) {
    (void)in_sizes; (void)n_in; (void)out_size; (void)ws_size;
    const int M = CB * CT;                     // 4096 rows

    const int*   idx    = (const int*)  d_in[0];
    const float* q_emb  = (const float*)d_in[1];
    const float* p_emb  = (const float*)d_in[2];
    const float* ln1_w  = (const float*)d_in[3];
    const float* ln1_b  = (const float*)d_in[4];
    const float* attn_w = (const float*)d_in[5];
    const float* proj_w = (const float*)d_in[6];
    const float* grav_w = (const float*)d_in[7];
    const float* grav_b = (const float*)d_in[8];
    const float* conv_w = (const float*)d_in[9];
    const float* conv_b = (const float*)d_in[10];
    const float* ln2_w  = (const float*)d_in[11];
    const float* ln2_b  = (const float*)d_in[12];
    const float* mlp_w1 = (const float*)d_in[13];
    const float* mlp_w2 = (const float*)d_in[14];
    const float* lnf_w  = (const float*)d_in[15];
    const float* lnf_b  = (const float*)d_in[16];

    // ---- workspace carve-out (deterministic) ----
    char* wp = (char*)d_ws;
    auto alloc = [&](size_t bytes) -> char* {
        char* p = wp;
        wp += (bytes + 255) & ~(size_t)255;
        return p;
    };
    float* x      = (float*)alloc((size_t)M * CC * 4);
    bf16*  xb     = (bf16*) alloc((size_t)M * CC * 2);
    float* qkv    = (float*)alloc((size_t)M * 3 * CC * 4);
    bf16*  qb     = (bf16*) alloc((size_t)M * CC * 2);
    bf16*  kb     = (bf16*) alloc((size_t)M * CC * 2);
    bf16*  vtb    = (bf16*) alloc((size_t)M * CC * 2);
    float* gbuf   = (float*)alloc((size_t)CB * CH * CT * 4);
    bf16*  yb     = (bf16*) alloc((size_t)M * CC * 2);
    float* projo  = (float*)alloc((size_t)M * CC * 4);
    float* reflex = (float*)alloc((size_t)M * CC * 4);
    bf16*  mact   = (bf16*) alloc((size_t)M * 4 * CC * 2);
    float* mlpo   = (float*)alloc((size_t)M * CC * 4);
    bf16*  wattn  = (bf16*) alloc((size_t)CL * 3 * CC * CC * 2);
    bf16*  wproj  = (bf16*) alloc((size_t)CL * CC * CC * 2);
    bf16*  wm1    = (bf16*) alloc((size_t)CL * 4 * CC * CC * 2);
    bf16*  wm2    = (bf16*) alloc((size_t)CL * CC * 4 * CC * 2);
    bf16*  wemb   = (bf16*) alloc((size_t)CV * CC * 2);

    auto cvt = [&](const float* src, bf16* dst, size_t n) {
        cvt_bf16_k<<<(unsigned)((n + 255) / 256), 256, 0, stream>>>(src, dst, n);
    };
    cvt(attn_w, wattn, (size_t)CL * 3 * CC * CC);
    cvt(proj_w, wproj, (size_t)CL * CC * CC);
    cvt(mlp_w1, wm1,   (size_t)CL * 4 * CC * CC);
    cvt(mlp_w2, wm2,   (size_t)CL * CC * 4 * CC);
    cvt(q_emb,  wemb,  (size_t)CV * CC);

    embed_ema_k<<<(CB * CC + 255) / 256, 256, 0, stream>>>(idx, q_emb, p_emb, x);

    const size_t nBTC = (size_t)M * CC;
    const unsigned gEl = (unsigned)((nBTC + 255) / 256);
    const unsigned gM = (unsigned)((M + 255) / 256);      // gemm grid.x (256-row blocks)

    for (int l = 0; l < CL; ++l) {
        conv_silu_k<<<gEl, 256, 0, stream>>>(x, conv_w + (size_t)l * CC * CK,
                                             conv_b + (size_t)l * CC, reflex);
        layernorm_k<false><<<M, 256, 0, stream>>>(x, ln1_w + (size_t)l * CC,
                                                  ln1_b + (size_t)l * CC, xb, nullptr);
        gemm_bf16_k<0, false><<<dim3(gM, (3 * CC + 63) / 64), 256, 0, stream>>>(
            xb, wattn + (size_t)l * 3 * CC * CC, qkv, nullptr, M, 3 * CC, CC);
        rope_split_k<<<gEl, 256, 0, stream>>>(qkv, qb, kb, vtb);
        gravity_k<<<(M * CH + 255) / 256, 256, 0, stream>>>(
            xb, grav_w + (size_t)l * CH * CC, grav_b + (size_t)l * CH, gbuf);
        flash_attn_k<<<CB * CH * (CT / 64), 128, 0, stream>>>(qb, kb, vtb, gbuf, yb, 0.125f);
        gemm_bf16_k<0, false><<<dim3(gM, (CC + 63) / 64), 256, 0, stream>>>(
            yb, wproj + (size_t)l * CC * CC, projo, nullptr, M, CC, CC);
        add3_k<<<gEl, 256, 0, stream>>>(x, projo, reflex, nBTC);
        layernorm_k<false><<<M, 256, 0, stream>>>(x, ln2_w + (size_t)l * CC,
                                                  ln2_b + (size_t)l * CC, xb, nullptr);
        gemm_bf16_k<1, true><<<dim3(gM, (4 * CC + 63) / 64), 256, 0, stream>>>(
            xb, wm1 + (size_t)l * 4 * CC * CC, nullptr, mact, M, 4 * CC, CC);
        gemm_bf16_k<0, false><<<dim3(gM, (CC + 63) / 64), 256, 0, stream>>>(
            mact, wm2 + (size_t)l * CC * 4 * CC, mlpo, nullptr, M, CC, 4 * CC);
        add1_k<<<gEl, 256, 0, stream>>>(x, mlpo, nBTC);
    }

    float* out_logits = (float*)d_out;                   // [B,T,V] fp32
    float* out_x = out_logits + (size_t)M * CV;          // [B,T,C] fp32
    layernorm_k<true><<<M, 256, 0, stream>>>(x, lnf_w, lnf_b, xb, out_x);
    gemm_bf16_k<0, false><<<dim3(gM, (CV + 63) / 64), 256, 0, stream>>>(
        xb, wemb, out_logits, nullptr, M, CV, CC);
}
